// BasicRecurrentEntityEncoder_44530220925018
// MI455X (gfx1250) — compile-verified, matrix operand order checked by reading, not run on hardware
//
#include <hip/hip_runtime.h>

// Problem constants (from reference)
constexpr int B_ = 64, T_ = 128, L_ = 32, D_ = 128, K_ = 64;

typedef __attribute__((ext_vector_type(16))) __bf16 v16bf;
typedef __attribute__((ext_vector_type(8)))  float  v8f;

#define DEV __device__ __forceinline__

// f32 -> bf16 round-to-nearest-even, via bit ops (no fptrunc dependence)
DEV __bf16 f2bf(float x) {
  unsigned u = __builtin_bit_cast(unsigned, x);
  unsigned r = (u + 0x7FFFu + ((u >> 16) & 1u)) >> 16;
  unsigned short s = (unsigned short)r;
  return __builtin_bit_cast(__bf16, s);
}

// Permute K within each 32-block so a lane's WMMA fragment is 32 contiguous
// bytes in LDS: order [0..7, 16..23, 8..15, 24..31]  (matches the ISA 16-bit
// A/B layout: lanes 0-15 hold K 0-7/16-23, lanes 16-31 hold K 8-15/24-31).
DEV int permk(int k) {
  int j = k & 31;
  int p = (j & 7) | (((j >> 4) & 1) << 3) | (((j >> 3) & 1) << 4);
  return (k & ~31) | p;
}

DEV float sigmoidf(float x) { return 1.0f / (1.0f + __expf(-x)); }

// ---------------------------------------------------------------------------
// Kernel 1: enc[b,t,d] = sum_l E[idx[b,t,l],d] * mask[b,t,l] * pos[l,d]
//           smask[b,t] = any_l mask[b,t,l]
// One block per (b,t), 128 threads (one per d). Vocab table is L2-resident.
// ---------------------------------------------------------------------------
__global__ void __launch_bounds__(128)
encode_kernel(const int* __restrict__ prgrph, const unsigned char* __restrict__ pmask,
              const float* __restrict__ emb, const float* __restrict__ pos,
              float* __restrict__ enc, int* __restrict__ smask)
{
  const int bt = blockIdx.x;
  const int d  = threadIdx.x;
  const int* idx = prgrph + bt * L_;
  const unsigned char* mk = pmask + bt * L_;
  float acc = 0.0f;
  int any = 0;
  #pragma unroll 4
  for (int l = 0; l < L_; ++l) {
    int m = mk[l];
    any |= m;
    if (m) acc += emb[(long)idx[l] * D_ + d] * pos[l * D_ + d];
  }
  enc[(long)bt * D_ + d] = acc;
  if (d == 0) smask[bt] = any;
}

// ---------------------------------------------------------------------------
// Kernel 2: Out = A @ W  ((M x 128) @ (128 x 128)), f32 in/out, bf16 WMMA
// compute with f32 accumulation. 64 rows per block, 8 waves; wave w owns
// N-tile w, sweeps 4 M-tiles x 4 K-chunks (B fragments stay in registers).
// Used for SW = ENC @ Ww  and  KV = KEYS @ Vw.
// ---------------------------------------------------------------------------
__global__ void __launch_bounds__(256)
gemm128_bf16_kernel(const float* __restrict__ A, const float* __restrict__ W,
                    float* __restrict__ Out)
{
  __shared__ __align__(32) __bf16 Abf[64 * 128];   // K-permuted rows of A
  __shared__ __align__(32) __bf16 WT [128 * 128];  // W transposed, K-permuted

  const int tid   = threadIdx.x;
  const int lane  = tid & 31;
  const int wv    = tid >> 5;
  const int rowl  = lane & 15;
  const int khalf = lane >> 4;
  const long rowbase = (long)blockIdx.x * 64;

  for (int i = tid; i < 128 * 128; i += 256) {
    int k = i >> 7, n = i & 127;
    WT[n * 128 + permk(k)] = f2bf(W[i]);          // W is [k][n] row-major
  }
  for (int i = tid; i < 64 * 128; i += 256) {
    int m = i >> 7, k = i & 127;
    Abf[m * 128 + permk(k)] = f2bf(A[(rowbase + m) * 128 + k]);
  }
  __syncthreads();

  v16bf bfrag[4];
  #pragma unroll
  for (int kc = 0; kc < 4; ++kc)
    bfrag[kc] = *(const v16bf*)&WT[(wv * 16 + rowl) * 128 + kc * 32 + khalf * 16];

  #pragma unroll
  for (int tm = 0; tm < 4; ++tm) {
    v8f acc = {};
    #pragma unroll
    for (int kc = 0; kc < 4; ++kc) {
      v16bf a = *(const v16bf*)&Abf[(tm * 16 + rowl) * 128 + kc * 32 + khalf * 16];
      acc = __builtin_amdgcn_wmma_f32_16x16x32_bf16(false, a, false, bfrag[kc],
                                                    (short)0, acc, false, false);
    }
    const int n = wv * 16 + rowl;
    #pragma unroll
    for (int r = 0; r < 8; ++r) {
      int m = tm * 16 + r + khalf * 8;             // C layout: lanes16-31 -> M+8
      Out[(rowbase + m) * 128 + n] = acc[r];
    }
  }
}

// ---------------------------------------------------------------------------
// Kernel 3: sequential entity scan. 256 blocks (64 b x 4 tiles of 16 entities),
// 256 threads = 8 wave32.  Per-thread-private state (h, key, kv C-tile) lives
// in registers for all 128 steps; Uw B-fragments are hoisted into VGPRs; gate
// and L2-norm reductions use wave shuffles.  Only 2 barriers per step:
//   P1 (regs -> Hbf, gate via shfl)  |bar|  P2 (4x WMMA -> Htil)  |bar|
//   P4 (gated update + norm via shfl, commit to regs)
// ---------------------------------------------------------------------------
__global__ void __launch_bounds__(256)
entity_scan_kernel(const float* __restrict__ enc, const float* __restrict__ sw,
                   const float* __restrict__ kv, const float* __restrict__ keys,
                   const float* __restrict__ Uw, const int* __restrict__ smask,
                   float* __restrict__ out)
{
  __shared__ __align__(32) __bf16 UwT [128 * 128]; // B-frag staging (init only)
  __shared__ __align__(32) float  Htil[16 * 128];  // sigmoid(H@Uw + KV + SW)
  __shared__ __align__(32) __bf16 Hbf [16 * 128];  // bf16 H, K-permuted

  const int tid   = threadIdx.x;
  const int b     = blockIdx.x >> 2;
  const int ke0   = (blockIdx.x & 3) * 16;
  const int lane  = tid & 31;
  const int wv    = tid >> 5;
  const int rowl  = lane & 15;
  const int khalf = lane >> 4;
  const int eg    = tid >> 4;          // entity owned in elementwise phases
  const int d0    = (tid & 15) * 8;    // dim base of this thread's chunk
  const int i0    = eg * D_ + d0;      // == tid*8
  const int ncol  = wv * 16 + rowl;    // this lane's output column in P2

  // ---- loop-invariant per-thread registers -------------------------------
  float hreg[8], keyreg[8], kvreg[8];
  #pragma unroll
  for (int j = 0; j < 8; ++j) {
    hreg[j]   = 0.0f;
    keyreg[j] = keys[(long)(b * K_ + ke0 + eg) * D_ + d0 + j];
  }
  #pragma unroll
  for (int r = 0; r < 8; ++r)          // lane's C-tile bias KV[m][ncol]
    kvreg[r] = kv[(long)(b * K_ + ke0 + r + khalf * 8) * D_ + ncol];

  // ---- stage Uw (transposed, K-permuted, bf16) and hoist B fragments -----
  for (int i = tid; i < D_ * D_; i += 256) {
    int n = i >> 7, k = i & 127;
    UwT[n * D_ + permk(k)] = f2bf(Uw[k * D_ + n]);
  }
  __syncthreads();
  v16bf bfrag[4];
  #pragma unroll
  for (int kc = 0; kc < 4; ++kc)
    bfrag[kc] = *(const v16bf*)&UwT[ncol * D_ + kc * 32 + khalf * 16];

  __bf16* hbp = &Hbf[eg * D_ + permk(d0)];   // own chunk, contiguous after perm

  for (int t = 0; t < T_; ++t) {
    if (!smask[b * T_ + t]) continue;        // uniform per block: skip step
    const float* sp = enc + (long)(b * T_ + t) * D_;

    // P1: publish h as bf16; gate g = sigmoid((h+key).s) via wave shuffle
    #pragma unroll
    for (int j = 0; j < 8; ++j) hbp[j] = f2bf(hreg[j]);
    float gp = 0.0f;
    #pragma unroll
    for (int j = 0; j < 8; ++j) gp += (hreg[j] + keyreg[j]) * sp[d0 + j];
    gp += __shfl_xor(gp, 1);
    gp += __shfl_xor(gp, 2);
    gp += __shfl_xor(gp, 4);
    gp += __shfl_xor(gp, 8);
    const float g    = sigmoidf(gp);
    const float sw_n = sw[(long)(b * T_ + t) * D_ + ncol];
    __syncthreads();

    // P2: Htil = sigmoid(H @ Uw + KV + SW); wave wv owns N-tile wv
    {
      v8f acc = {};
      #pragma unroll
      for (int kc = 0; kc < 4; ++kc) {
        v16bf a = *(const v16bf*)&Hbf[rowl * D_ + kc * 32 + khalf * 16];
        acc = __builtin_amdgcn_wmma_f32_16x16x32_bf16(false, a, false, bfrag[kc],
                                                      (short)0, acc, false, false);
      }
      #pragma unroll
      for (int r = 0; r < 8; ++r) {
        int m = r + khalf * 8;               // C layout: lanes16-31 -> M+8
        Htil[m * D_ + ncol] = sigmoidf(acc[r] + kvreg[r] + sw_n);
      }
    }
    __syncthreads();

    // P4: hn = h + g*h_tilda; L2-normalize (wave shuffle); commit to regs.
    // (h is thread-private: next P1 re-reads only this thread's chunk, so no
    //  trailing barrier is needed.)
    {
      float hn[8], ss = 0.0f;
      #pragma unroll
      for (int j = 0; j < 8; ++j) {
        hn[j] = hreg[j] + g * Htil[i0 + j];
        ss += hn[j] * hn[j];
      }
      ss += __shfl_xor(ss, 1);
      ss += __shfl_xor(ss, 2);
      ss += __shfl_xor(ss, 4);
      ss += __shfl_xor(ss, 8);
      const float rn = rsqrtf(fmaxf(ss, 1e-12f));
      #pragma unroll
      for (int j = 0; j < 8; ++j) hreg[j] = hn[j] * rn;
    }
  }

  float* op = out + (long)(b * K_ + ke0 + eg) * D_ + d0;
  #pragma unroll
  for (int j = 0; j < 8; ++j) op[j] = hreg[j];
}

// ---------------------------------------------------------------------------
extern "C" void kernel_launch(void* const* d_in, const int* in_sizes, int n_in,
                              void* d_out, int out_size, void* d_ws, size_t ws_size,
                              hipStream_t stream) {
  const int*            prgrph = (const int*)d_in[0];
  const unsigned char*  pmask  = (const unsigned char*)d_in[1]; // jnp bool = 1 byte
  const float*          emb    = (const float*)d_in[2];
  const float*          pos    = (const float*)d_in[3];
  const float*          Uw     = (const float*)d_in[4];
  const float*          Vw     = (const float*)d_in[5];
  const float*          Ww     = (const float*)d_in[6];
  const float*          keys   = (const float*)d_in[7];
  float*                out    = (float*)d_out;

  // Workspace partition (all f32): enc[B,T,D], SW[B,T,D], KV[B,K,D], smask[B,T]
  float* enc   = (float*)d_ws;
  float* swv   = enc + (size_t)B_ * T_ * D_;
  float* kvv   = swv + (size_t)B_ * T_ * D_;
  int*   smask = (int*)(kvv + (size_t)B_ * K_ * D_);

  // 1) Sentence encode (gather-reduce; vocab table L2-resident)
  encode_kernel<<<B_ * T_, 128, 0, stream>>>(prgrph, pmask, emb, pos, enc, smask);

  // 2) Loop-invariant matmuls hoisted out of the scan (bf16 WMMA)
  gemm128_bf16_kernel<<<(B_ * T_) / 64, 256, 0, stream>>>(enc, Ww, swv);   // SW = enc @ Ww
  gemm128_bf16_kernel<<<(B_ * K_) / 64, 256, 0, stream>>>(keys, Vw, kvv);  // KV = keys @ Vw

  // 3) Sequential scan: 256 blocks (64 b x 4 entity tiles), register-resident
  //    state, 2 barriers/step
  entity_scan_kernel<<<B_ * (K_ / 16), 256, 0, stream>>>(
      enc, swv, kvv, keys, Uw, smask, out);
}
